// CasualSelfAttention_5523327942949
// MI455X (gfx1250) — matrix operand order repaired and checked by reading, not compiled, not run on hardware
//
#include <hip/hip_runtime.h>

// Causal self-attention forward for MI455X (gfx1250), wave32 + WMMA bf16.
// B=4, T=2048, C=1024, H=16, hd=64.

static constexpr int Bn = 4;
static constexpr int Tn = 2048;
static constexpr int Cn = 1024;
static constexpr int Hn = 16;
static constexpr int HD = 64;

typedef __bf16 bf16;
typedef __attribute__((ext_vector_type(8)))  bf16  v8bf;
typedef __attribute__((ext_vector_type(16))) bf16  v16bf;
typedef __attribute__((ext_vector_type(8)))  float v8f;

// ---------------------------------------------------------------------------
// Fragment loaders per CDNA5 WMMA VGPR layouts (ISA 7.12.2), 16x16x32 bf16.
// A (16x32 MxK): lane l -> row m=l&15, half=l>>4.
//   elements e=0..7  -> k = half*8 + e          (16B contiguous)
//   elements e=8..15 -> k = 16 + half*8 + (e-8) (16B contiguous)
__device__ __forceinline__ v16bf frag_a(const bf16* rowk0, int half) {
  v8bf lo = *(const v8bf*)(rowk0 + half * 8);
  v8bf hi = *(const v8bf*)(rowk0 + 16 + half * 8);
  v16bf r;
#pragma unroll
  for (int i = 0; i < 8; ++i) { r[i] = lo[i]; r[i + 8] = hi[i]; }
  return r;
}

// B (32x16 KxN): lane l -> col n=l&15, half=l>>4; element e<8 -> k = half*8+e, e>=8 -> k = 16+half*8+(e-8).
// Source must be stored K-contiguous per column (i.e. "transposed").
__device__ __forceinline__ v16bf frag_b(const bf16* colk0, int half) {
  v8bf lo = *(const v8bf*)(colk0 + half * 8);
  v8bf hi = *(const v8bf*)(colk0 + 16 + half * 8);
  v16bf r;
#pragma unroll
  for (int i = 0; i < 8; ++i) { r[i] = lo[i]; r[i + 8] = hi[i]; }
  return r;
}

__device__ __forceinline__ v8f wmma_bf16(v16bf a, v16bf b, v8f c) {
  return __builtin_amdgcn_wmma_f32_16x16x32_bf16(false, a, false, b,
                                                 (short)0, c, false, false);
}

// ---------------------------------------------------------------------------
// fp32 -> bf16 elementwise
__global__ __launch_bounds__(256) void cvt_bf16_kernel(const float* __restrict__ in,
                                                       bf16* __restrict__ out, int n) {
  int i = blockIdx.x * 256 + threadIdx.x;
  if (i < n) out[i] = (bf16)in[i];
}

// fp32 [rows][cols] -> bf16 [cols][rows] (weight transpose so B-fragments are
// K-contiguous)
__global__ __launch_bounds__(256) void cvt_transpose_kernel(const float* __restrict__ in,
                                                            bf16* __restrict__ out,
                                                            int rows, int cols) {
  int i = blockIdx.x * 256 + threadIdx.x;
  if (i >= rows * cols) return;
  int r = i / cols, c = i - r * cols;
  out[(size_t)c * rows + r] = (bf16)in[i];
}

// ---------------------------------------------------------------------------
// qkv = x @ w_attn + b_attn, scattered into per-head Q[B,H,T,hd], K[B,H,T,hd],
// Vt[B,H,hd,T] (V transposed so the PV WMMA B-operand is key-contiguous).
// Block = 256 threads (8 waves); per-wave output tile 32(M) x 64(N) so each
// B fragment is reused across two A fragments (L2 fetch intensity ~22 FLOP/B).
// Block tile 64 x 256. Grid (BT/64, 3C/256).
__global__ __launch_bounds__(256) void qkv_gemm_kernel(
    const bf16* __restrict__ xb, const bf16* __restrict__ wAt,
    const float* __restrict__ bias, bf16* __restrict__ q, bf16* __restrict__ k,
    bf16* __restrict__ vT) {
  const int lane = threadIdx.x & 31;
  const int wid  = threadIdx.x >> 5;
  const int n    = lane & 15;
  const int half = lane >> 4;
  const int m0 = blockIdx.x * 64 + (wid >> 2) * 32;   // row in [0, B*T)
  const int n0 = blockIdx.y * 256 + (wid & 3) * 64;   // col in [0, 3C)

  v8f acc[2][4] = {};
  const bf16* arow0 = xb + (size_t)(m0 + n) * Cn;       // A rows for this lane
  const bf16* arow1 = xb + (size_t)(m0 + 16 + n) * Cn;
  for (int kt = 0; kt < Cn; kt += 32) {
    // locality=3 -> WGP-scope prefetch: pulls into all cache levels (ISA 10.5),
    // warming WGP$ for the next k-tile's b128 clauses (data is already L2-hot).
    __builtin_prefetch(arow0 + kt + 64, 0, 3);
    __builtin_prefetch(arow1 + kt + 64, 0, 3);
    v16bf a0 = frag_a(arow0 + kt, half);
    v16bf a1 = frag_a(arow1 + kt, half);
#pragma unroll
    for (int j = 0; j < 4; ++j) {
      v16bf b = frag_b(wAt + (size_t)(n0 + j * 16 + n) * Cn + kt, half);
      acc[0][j] = wmma_bf16(a0, b, acc[0][j]);
      acc[1][j] = wmma_bf16(a1, b, acc[1][j]);
    }
  }

#pragma unroll
  for (int j = 0; j < 4; ++j) {
    const int jc = n0 + j * 16 + n;
    const float bv = bias[jc];
    const int which = jc >> 10;       // 0=Q 1=K 2=V
    const int jj = jc & (Cn - 1);
    const int h = jj >> 6, d = jj & (HD - 1);
#pragma unroll
    for (int i = 0; i < 2; ++i) {
#pragma unroll
      for (int r = 0; r < 8; ++r) {
        const int row = m0 + i * 16 + r + 8 * half;
        const int bb = row >> 11, tt = row & (Tn - 1);
        const size_t hb = (size_t)bb * Hn + h;
        const bf16 val = (bf16)(acc[i][j][r] + bv);
        if (which == 0)      q[(hb * Tn + tt) * HD + d] = val;
        else if (which == 1) k[(hb * Tn + tt) * HD + d] = val;
        else                 vT[(hb * HD + d) * Tn + tt] = val;
      }
    }
  }
}

// ---------------------------------------------------------------------------
// Flash attention: one wave owns a 16-query tile; streams 32-key chunks with
// online softmax. Per chunk: 4 QK WMMAs + 4 PV WMMAs. Probs bounce through a
// per-wave LDS tile to rotate from C/D layout to A layout (DS in-order within
// a wave; explicit s_wait_dscnt, no cross-wave sync -> divergent trip counts OK).
__global__ __launch_bounds__(128) void attn_kernel(
    const bf16* __restrict__ q, const bf16* __restrict__ k,
    const bf16* __restrict__ vT, bf16* __restrict__ yb) {
  const int lane = threadIdx.x & 31;
  const int wid  = threadIdx.x >> 5;
  const int n    = lane & 15;
  const int half = lane >> 4;
  const int qt = blockIdx.x * 4 + wid;    // query tile within T
  const int bh = blockIdx.y;              // b*H + h
  const int bb = bh >> 4, h = bh & 15;

  const bf16* qh = q  + (size_t)bh * Tn * HD;
  const bf16* kh = k  + (size_t)bh * Tn * HD;
  const bf16* vh = vT + (size_t)bh * HD * Tn;

  __shared__ __align__(16) bf16 pbuf[4][16][40];  // 40: pad rows to spread banks

  const int m0 = qt * 16;
  const bf16* qrow = qh + (size_t)(m0 + n) * HD;  // A row (m = lane&15)
  const v16bf aq0 = frag_a(qrow, half);
  const v16bf aq1 = frag_a(qrow + 32, half);

  v8f o[4] = {};
  float rm[8], rs[8];
#pragma unroll
  for (int r = 0; r < 8; ++r) { rm[r] = -1e30f; rs[r] = 0.0f; }
  const float scale = 0.125f;  // 1/sqrt(64)
  const int kend = m0 + 15;    // causal: last needed key

  for (int k0 = 0; k0 <= kend; k0 += 32) {
    // S = Q K^T for two 16-key subtiles
    v8f s0 = {}, s1 = {};
    {
      const bf16* kb0 = kh + (size_t)(k0 + n) * HD;
      s0 = wmma_bf16(aq0, frag_b(kb0, half), s0);
      s0 = wmma_bf16(aq1, frag_b(kb0 + 32, half), s0);
      const bf16* kb1 = kh + (size_t)(k0 + 16 + n) * HD;
      s1 = wmma_bf16(aq0, frag_b(kb1, half), s1);
      s1 = wmma_bf16(aq1, frag_b(kb1 + 32, half), s1);
    }

    // Online softmax update (per row r; 16 lanes of a half-group share a row)
#pragma unroll
    for (int r = 0; r < 8; ++r) {
      const int qrow_g = m0 + r + 8 * half;
      const float a0 = (k0 + n      <= qrow_g) ? s0[r] * scale : -1e30f;
      const float a1 = (k0 + 16 + n <= qrow_g) ? s1[r] * scale : -1e30f;
      float mx = fmaxf(a0, a1);
#pragma unroll
      for (int off = 1; off < 16; off <<= 1)
        mx = fmaxf(mx, __shfl_xor(mx, off, 32));
      const float mnew = fmaxf(rm[r], mx);
      const float corr = __expf(rm[r] - mnew);
      rm[r] = mnew;
      const float p0 = __expf(a0 - mnew);
      const float p1 = __expf(a1 - mnew);
      float psum = p0 + p1;
#pragma unroll
      for (int off = 1; off < 16; off <<= 1)
        psum += __shfl_xor(psum, off, 32);
      rs[r] = rs[r] * corr + psum;
#pragma unroll
      for (int j = 0; j < 4; ++j) o[j][r] = o[j][r] * corr;
      pbuf[wid][r + 8 * half][n]      = (bf16)p0;
      pbuf[wid][r + 8 * half][16 + n] = (bf16)p1;
    }

    asm volatile("s_wait_dscnt 0" ::: "memory");  // DS in-order per wave

    // O += P V : A = probs (16x32), B = Vt (key-contiguous per hd column)
    const v16bf ap = frag_a(&pbuf[wid][n][0], half);
#pragma unroll
    for (int j = 0; j < 4; ++j) {
      const bf16* vb = vh + (size_t)(j * 16 + n) * Tn + k0;
      o[j] = wmma_bf16(ap, frag_b(vb, half), o[j]);
    }
  }

  // Normalize and store y[B,T,C] (bf16 input to projection GEMM)
  float inv[8];
#pragma unroll
  for (int r = 0; r < 8; ++r) inv[r] = 1.0f / rs[r];
#pragma unroll
  for (int j = 0; j < 4; ++j) {
#pragma unroll
    for (int r = 0; r < 8; ++r) {
      const size_t row = (size_t)bb * Tn + m0 + r + 8 * half;
      yb[row * Cn + h * HD + j * 16 + n] = (bf16)(o[j][r] * inv[r]);
    }
  }
}

// ---------------------------------------------------------------------------
// out = y @ w_proj + b_proj (fp32 output). Same 32x64-per-wave tiling.
__global__ __launch_bounds__(256) void proj_gemm_kernel(
    const bf16* __restrict__ yb, const bf16* __restrict__ wPt,
    const float* __restrict__ bias, float* __restrict__ out) {
  const int lane = threadIdx.x & 31;
  const int wid  = threadIdx.x >> 5;
  const int n    = lane & 15;
  const int half = lane >> 4;
  const int m0 = blockIdx.x * 64 + (wid >> 2) * 32;
  const int n0 = blockIdx.y * 256 + (wid & 3) * 64;

  v8f acc[2][4] = {};
  const bf16* arow0 = yb + (size_t)(m0 + n) * Cn;
  const bf16* arow1 = yb + (size_t)(m0 + 16 + n) * Cn;
  for (int kt = 0; kt < Cn; kt += 32) {
    __builtin_prefetch(arow0 + kt + 64, 0, 3);
    __builtin_prefetch(arow1 + kt + 64, 0, 3);
    v16bf a0 = frag_a(arow0 + kt, half);
    v16bf a1 = frag_a(arow1 + kt, half);
#pragma unroll
    for (int j = 0; j < 4; ++j) {
      v16bf b = frag_b(wPt + (size_t)(n0 + j * 16 + n) * Cn + kt, half);
      acc[0][j] = wmma_bf16(a0, b, acc[0][j]);
      acc[1][j] = wmma_bf16(a1, b, acc[1][j]);
    }
  }
#pragma unroll
  for (int j = 0; j < 4; ++j) {
    const int jc = n0 + j * 16 + n;
    const float bv = bias[jc];
#pragma unroll
    for (int i = 0; i < 2; ++i) {
#pragma unroll
      for (int r = 0; r < 8; ++r) {
        const size_t row = (size_t)m0 + i * 16 + r + 8 * half;
        out[row * Cn + jc] = acc[i][j][r] + bv;
      }
    }
  }
}

// ---------------------------------------------------------------------------
extern "C" void kernel_launch(void* const* d_in, const int* in_sizes, int n_in,
                              void* d_out, int out_size, void* d_ws, size_t ws_size,
                              hipStream_t stream) {
  const float* x      = (const float*)d_in[0];
  const float* w_attn = (const float*)d_in[1];
  const float* b_attn = (const float*)d_in[2];
  const float* w_proj = (const float*)d_in[3];
  const float* b_proj = (const float*)d_in[4];
  float* out = (float*)d_out;

  char* ws = (char*)d_ws;
  const size_t MB = 1024ull * 1024ull;
  bf16* xb  = (bf16*)(ws + 0 * MB);   // [B*T, C]      16 MB
  bf16* wAt = (bf16*)(ws + 16 * MB);  // [3C, C]        6 MB
  bf16* wPt = (bf16*)(ws + 22 * MB);  // [C, C]         2 MB
  bf16* q   = (bf16*)(ws + 24 * MB);  // [B,H,T,hd]    16 MB
  bf16* k   = (bf16*)(ws + 40 * MB);  // [B,H,T,hd]    16 MB
  bf16* vT  = (bf16*)(ws + 56 * MB);  // [B,H,hd,T]    16 MB
  bf16* yb  = (bf16*)(ws + 72 * MB);  // [B*T, C]      16 MB  (total 88 MB)

  const int nx = Bn * Tn * Cn;
  cvt_bf16_kernel<<<nx / 256, 256, 0, stream>>>(x, xb, nx);
  cvt_transpose_kernel<<<(Cn * 3 * Cn) / 256, 256, 0, stream>>>(w_attn, wAt, Cn, 3 * Cn);
  cvt_transpose_kernel<<<(Cn * Cn) / 256, 256, 0, stream>>>(w_proj, wPt, Cn, Cn);

  qkv_gemm_kernel<<<dim3(Bn * Tn / 64, 3 * Cn / 256), 256, 0, stream>>>(
      xb, wAt, b_attn, q, k, vT);

  attn_kernel<<<dim3(Tn / 64, Bn * Hn), 128, 0, stream>>>(q, k, vT, yb);

  proj_gemm_kernel<<<dim3(Bn * Tn / 64, Cn / 256), 256, 0, stream>>>(
      yb, wPt, b_proj, out);
}
